// LoRA_89326729822539
// MI455X (gfx1250) — matrix operand order repaired and checked by reading, not compiled, not run on hardware
//
#include <hip/hip_runtime.h>
#include <hip/hip_bf16.h>

typedef __attribute__((ext_vector_type(16))) __bf16 v16bf;
typedef __attribute__((ext_vector_type(8)))  __bf16 v8bf;
typedef __attribute__((ext_vector_type(8)))  float  v8f;
typedef __attribute__((ext_vector_type(4)))  int    v4i;

typedef __attribute__((address_space(1))) v4i g_v4i;   // global
typedef __attribute__((address_space(3))) v4i l_v4i;   // LDS

#define K_IN   4096
#define D_OUT  4096
#define N_COL  8192
#define R_RANK 16
#define SCALER 2.0f   // ALPHA / R = 32/16

#define TILE_M 128
#define TILE_N 256
#define TILE_K 64
#define LDSS   72     // padded LDS row stride in bf16 elements (64 + 8)
#define NKCH   (K_IN / TILE_K)

static __device__ __forceinline__ v16bf concat8(v8bf a, v8bf b) {
    return __builtin_shufflevector(a, b, 0,1,2,3,4,5,6,7,8,9,10,11,12,13,14,15);
}

// ---- CDNA5 async global->LDS copy (ASYNCcnt path) --------------------------
static __device__ __forceinline__ void async_copy16(const __bf16* g, __bf16* l) {
#if __has_builtin(__builtin_amdgcn_global_load_async_to_lds_b128)
    // Signature (from compiler diagnostic): (v4i AS1*, v4i AS3*, imm offset, cpol).
    // Flat->global: address value identical. Flat->LDS: low 32 bits are the
    // LDS byte offset.
    g_v4i* gp = (g_v4i*)(unsigned long long)g;
    l_v4i* lp = (l_v4i*)(unsigned)(unsigned long long)l;
    __builtin_amdgcn_global_load_async_to_lds_b128(gp, lp, 0, 0);
#else
    unsigned lo = (unsigned)(unsigned long long)l;   // LDS byte address
    asm volatile("global_load_async_to_lds_b128 %0, %1, off"
                 :: "v"(lo), "v"(g) : "memory");
#endif
}

static __device__ __forceinline__ void wait_async_12() {
#if __has_builtin(__builtin_amdgcn_s_wait_asynccnt)
    __builtin_amdgcn_s_wait_asynccnt(12);
#else
    asm volatile("s_wait_asynccnt 12" ::: "memory");
#endif
}
static __device__ __forceinline__ void wait_async_0() {
#if __has_builtin(__builtin_amdgcn_s_wait_asynccnt)
    __builtin_amdgcn_s_wait_asynccnt(0);
#else
    asm volatile("s_wait_asynccnt 0" ::: "memory");
#endif
}

// ---------------------------------------------------------------------------
// Kernel 1: fold LoRA into weights, convert to bf16.
// Weff[m][k] = bf16( W0[m][k] + 2 * sum_r B[m][r] * A[r][k] )
// ---------------------------------------------------------------------------
__global__ __launch_bounds__(256) void lora_fold_weights(
    const float* __restrict__ W0, const float* __restrict__ A,
    const float* __restrict__ B, __bf16* __restrict__ Weff)
{
    size_t idx = (size_t)blockIdx.x * 256 + threadIdx.x;   // m*K_IN + k
    int k = (int)(idx & (K_IN - 1));
    int m = (int)(idx >> 12);                              // /4096
    float acc = 0.0f;
#pragma unroll
    for (int r = 0; r < R_RANK; ++r)
        acc += B[m * R_RANK + r] * A[(size_t)r * K_IN + k];
    Weff[idx] = (__bf16)(W0[idx] + SCALER * acc);
}

// ---------------------------------------------------------------------------
// Kernel 2: transpose + convert x (k-major f32) -> xT[n][k] bf16.
// ---------------------------------------------------------------------------
__global__ __launch_bounds__(256) void lora_transpose_x(
    const float* __restrict__ X, __bf16* __restrict__ Xt)
{
    __shared__ float tile[64][65];
    const int n0 = blockIdx.x * 64;
    const int k0 = blockIdx.y * 64;
    const int tx = threadIdx.x & 63;
    const int ty = threadIdx.x >> 6;      // 0..3
#pragma unroll
    for (int i = 0; i < 64; i += 4)
        tile[ty + i][tx] = X[(size_t)(k0 + ty + i) * N_COL + n0 + tx];
    __syncthreads();
#pragma unroll
    for (int i = 0; i < 64; i += 4)
        Xt[(size_t)(n0 + ty + i) * K_IN + k0 + tx] = (__bf16)tile[tx][ty + i];
}

// ---------------------------------------------------------------------------
// Kernel 3: H[m][n] = sum_k Weff[m][k] * xT[n][k]
// 256 threads = 8 waves (2 M x 4 N). WG tile 128x256, K-chunk 64,
// double-buffered LDS fed by async global->LDS copies (ASYNCcnt).
// Each wave: 64x64 = 4x4 v8f accumulators, v_wmma_f32_16x16x32_bf16.
// ---------------------------------------------------------------------------
__global__ __launch_bounds__(256) void lora_gemm_bf16(
    const __bf16* __restrict__ Weff,   // [D_OUT][K_IN]
    const __bf16* __restrict__ Xt,     // [N_COL][K_IN]
    float* __restrict__ H)             // [D_OUT][N_COL]
{
    __shared__ __align__(16) __bf16 sW[2][TILE_M * LDSS];   // 2 x 18 KB
    __shared__ __align__(16) __bf16 sX[2][TILE_N * LDSS];   // 2 x 36 KB

    const int t    = threadIdx.x;
    const int lane = t & 31;
    const int wave = t >> 5;
    const int wm   = wave >> 2;        // 0..1 -> 64-row slab
    const int wn   = wave & 3;         // 0..3 -> 64-col slab
    const int l16  = lane & 15;
    const int lh   = lane >> 4;        // lane half, selects K sub-group
    const int m0   = blockIdx.y * TILE_M;
    const int n0   = blockIdx.x * TILE_N;

    v8f acc[4][4];
#pragma unroll
    for (int i = 0; i < 4; ++i)
#pragma unroll
        for (int j = 0; j < 4; ++j)
            acc[i][j] = (v8f)(0.0f);

    // Per-chunk staging: sW = 1024 16B chunks (4/thread), sX = 2048 (8/thread)
    // => 12 async b128 ops per thread per K-chunk.
    auto stage = [&](int k0, int buf) {
#pragma unroll
        for (int i = 0; i < 4; ++i) {
            int c = t + i * 256, row = c >> 3, kc = (c & 7) * 8;
            async_copy16(&Weff[(size_t)(m0 + row) * K_IN + k0 + kc],
                         &sW[buf][row * LDSS + kc]);
        }
#pragma unroll
        for (int i = 0; i < 8; ++i) {
            int c = t + i * 256, row = c >> 3, kc = (c & 7) * 8;
            async_copy16(&Xt[(size_t)(n0 + row) * K_IN + k0 + kc],
                         &sX[buf][row * LDSS + kc]);
        }
    };

    auto compute = [&](int buf) {
#pragma unroll
        for (int kk = 0; kk < TILE_K; kk += 32) {
            v16bf afrag[4];
#pragma unroll
            for (int ti = 0; ti < 4; ++ti) {
                // A frag 16x32: lanes 0-15: K 0-7 & 16-23; lanes 16-31: K 8-15 & 24-31
                const __bf16* p =
                    &sW[buf][(wm * 64 + ti * 16 + l16) * LDSS + kk + lh * 8];
                afrag[ti] = concat8(*(const v8bf*)p, *(const v8bf*)(p + 16));
            }
#pragma unroll
            for (int tj = 0; tj < 4; ++tj) {
                // B frag 32x16: lanes 0-15: K 0-7 & 16-23; lanes 16-31: K 8-15 & 24-31
                const __bf16* p =
                    &sX[buf][(wn * 64 + tj * 16 + l16) * LDSS + kk + lh * 8];
                v16bf bfrag = concat8(*(const v8bf*)p, *(const v8bf*)(p + 16));
#pragma unroll
                for (int ti = 0; ti < 4; ++ti)
                    acc[ti][tj] = __builtin_amdgcn_wmma_f32_16x16x32_bf16(
                        false, afrag[ti], false, bfrag,
                        (short)0, acc[ti][tj], false, false);
            }
        }
    };

    stage(0, 0);
    for (int ik = 0; ik < NKCH - 1; ++ik) {
        const int cur = ik & 1;
        stage((ik + 1) * TILE_K, 1 - cur);  // target buffer free since barrier of iter ik-1
        wait_async_12();                    // chunk ik complete; chunk ik+1 (12 ops) pending
        __syncthreads();
        compute(cur);
        __syncthreads();                    // all reads of buf cur done before iter ik+1 overwrites
    }
    wait_async_0();
    __syncthreads();
    compute((NKCH - 1) & 1);

    // C/D layout: VGPR r, lanes 0-15 -> (M=r, N=lane); lanes 16-31 -> (M=r+8).
#pragma unroll
    for (int ti = 0; ti < 4; ++ti)
#pragma unroll
        for (int tj = 0; tj < 4; ++tj) {
            int mbase = m0 + wm * 64 + ti * 16 + lh * 8;
            int nbase = n0 + wn * 64 + tj * 16 + l16;
#pragma unroll
            for (int r = 0; r < 8; ++r)
                H[(size_t)(mbase + r) * N_COL + nbase] = acc[ti][tj][r];
        }
}

// ---------------------------------------------------------------------------
extern "C" void kernel_launch(void* const* d_in, const int* in_sizes, int n_in,
                              void* d_out, int out_size, void* d_ws, size_t ws_size,
                              hipStream_t stream) {
    const float* x  = (const float*)d_in[0];   // (4096, 8192)
    const float* W0 = (const float*)d_in[1];   // (4096, 4096)
    const float* A  = (const float*)d_in[2];   // (16, 4096)
    const float* B  = (const float*)d_in[3];   // (4096, 16)
    float* H = (float*)d_out;                  // (4096, 8192)

    __bf16* Weff = (__bf16*)d_ws;                                  // 32 MB
    __bf16* Xt   = (__bf16*)((char*)d_ws +
                   (size_t)D_OUT * K_IN * sizeof(__bf16));         // 64 MB

    lora_fold_weights<<<(D_OUT * (size_t)K_IN) / 256, 256, 0, stream>>>(W0, A, B, Weff);
    lora_transpose_x<<<dim3(N_COL / 64, K_IN / 64), 256, 0, stream>>>(x, Xt);
    lora_gemm_bf16<<<dim3(N_COL / TILE_N, D_OUT / TILE_M), 256, 0, stream>>>(Weff, Xt, H);
}
